// GAT_33337536151980
// MI455X (gfx1250) — matrix operand order repaired and checked by reading, not compiled, not run on hardware
//
#include <hip/hip_runtime.h>
#include <hip/hip_bf16.h>
#include <hip/hip_fp16.h>

typedef __attribute__((ext_vector_type(16))) _Float16 v16h;
typedef __attribute__((ext_vector_type(8)))  _Float16 v8h;
typedef __attribute__((ext_vector_type(8)))  float    v8f;

#define NNODES 100000
#define NEDGES 1600000
#define EPEDGES (NEDGES + NNODES)   // with self loops
#define FIN   256
#define NHEADS 8
#define HID   16
#define F1    (NHEADS * HID)        // 128
#define NCLS  40
#define C2PAD 48
#define NEG_SLOPE 0.2f

// ---- order-preserving float<->uint encoding for atomic max ----
__device__ __forceinline__ unsigned encf(float f) {
    unsigned u = __float_as_uint(f);
    return (u & 0x80000000u) ? ~u : (u | 0x80000000u);
}
__device__ __forceinline__ float decf(unsigned u) {
    return (u & 0x80000000u) ? __uint_as_float(u ^ 0x80000000u)
                             : __uint_as_float(~u);
}

// ---------------- prep: weights -> f16, pre-swizzled into WMMA B-fragment order
// B-frag element i of lane L, col-tile w, k-tile kt  <-  W[(kt*32 + (i>>3)*16 + (L>>4)*8 + (i&7))][w*16 + (L&15)]
// stored contiguously so each lane loads 16 halves (32B) with one vector load.
__global__ void gat_prep(const float* __restrict__ W1, const float* __restrict__ W2,
                         _Float16* __restrict__ W1s, _Float16* __restrict__ W2s) {
    int t = blockIdx.x * 256 + threadIdx.x;
    if (t < FIN * F1) {           // 8 col-tiles * 8 k-tiles * 32 lanes * 16 = 32768
        int i = t & 15, L = (t >> 4) & 31, kt = (t >> 9) & 7, w = t >> 12;
        int k = kt * 32 + (i >> 3) * 16 + (L >> 4) * 8 + (i & 7);
        int col = w * 16 + (L & 15);
        W1s[t] = (_Float16)W1[k * F1 + col];
    }
    if (t < 3 * 4 * 32 * 16) {    // 3 col-tiles * 4 k-tiles * 32 lanes * 16 = 6144
        int i = t & 15, L = (t >> 4) & 31, kt = (t >> 9) & 3, w = t >> 11;
        int k = kt * 32 + (i >> 3) * 16 + (L >> 4) * 8 + (i & 7);
        int col = w * 16 + (L & 15);
        W2s[t] = (col < NCLS) ? (_Float16)W2[k * NCLS + col] : (_Float16)0.f;
    }
}

__global__ void gat_fill_neginf(unsigned* __restrict__ p, int n) {
    int t = blockIdx.x * 256 + threadIdx.x;
    if (t < n) p[t] = encf(-1e30f);
}

// ---------------- layer-1 GEMM (WMMA f16 -> f32) ----------------------------
// block = 256 threads = 8 waves; wave w == head w (cols 16w..16w+15)
__global__ __launch_bounds__(256) void gat_gemm1(
    const float* __restrict__ x, const _Float16* __restrict__ W1s,
    _Float16* __restrict__ H1h) {
    __shared__ __align__(16) _Float16 xs[16][264];   // 528B row stride, 16B aligned
    int m0 = blockIdx.x * 16;
    int t = threadIdx.x;
    for (int i = 0; i < 16; ++i)
        xs[i][t] = (_Float16)x[(size_t)(m0 + i) * FIN + t];
    __syncthreads();

    int w = t >> 5, L = t & 31;
    int n = L & 15, half = L >> 4;
    int col = w * 16 + n;
    v8f c = {};
    #pragma unroll
    for (int kt = 0; kt < 8; ++kt) {
        int ab = kt * 32 + half * 8;                 // A: 16-bit 16x32 layout
        union { v16h v; v8h h[2]; } a;
        a.h[0] = *(const v8h*)&xs[n][ab];
        a.h[1] = *(const v8h*)&xs[n][ab + 16];
        v16h b = *(const v16h*)(W1s + (((w * 8 + kt) * 32 + L) << 4));
        c = __builtin_amdgcn_wmma_f32_16x16x32_f16(false, a.v, false, b, (short)0, c,
                                                   false, false);
    }
    #pragma unroll
    for (int r = 0; r < 8; ++r) {                    // C layout: M = r + 8*half
        int row = m0 + half * 8 + r;
        H1h[(size_t)row * F1 + col] = (_Float16)c[r];
    }
}

// ---------------- attention dots (memory-bound, no shuffles) ----------------
__global__ __launch_bounds__(256) void gat_att1(
    const _Float16* __restrict__ H1h,
    const float* __restrict__ att_s, const float* __restrict__ att_d,
    float* __restrict__ AS1, float* __restrict__ AD1) {
    int t = blockIdx.x * 256 + threadIdx.x;          // exactly N*8 threads
    int h = t & 7;
    const _Float16* hp = H1h + (size_t)t * HID;      // (nd*8+h)*16 == nd*128 + h*16
    float s = 0.f, d = 0.f;
    #pragma unroll
    for (int i = 0; i < HID; ++i) {
        float v = (float)hp[i];
        s += v * att_s[h * HID + i];
        d += v * att_d[h * HID + i];
    }
    AS1[t] = s; AD1[t] = d;
}

__global__ __launch_bounds__(256) void gat_att2(
    const _Float16* __restrict__ H2h,
    const float* __restrict__ att_s, const float* __restrict__ att_d,
    float* __restrict__ AS2, float* __restrict__ AD2) {
    int nd = blockIdx.x * 256 + threadIdx.x;
    if (nd >= NNODES) return;
    const _Float16* hp = H2h + (size_t)nd * C2PAD;
    float s = 0.f, d = 0.f;
    #pragma unroll
    for (int cc = 0; cc < NCLS; ++cc) {
        float v = (float)hp[cc];
        s += v * att_s[cc];
        d += v * att_d[cc];
    }
    AS2[nd] = s; AD2[nd] = d;
}

// ---------------- layer-1 edge passes ---------------------------------------
__global__ __launch_bounds__(256) void gat_edge_max1(
    const long long* __restrict__ ei,
    const float* __restrict__ AS1, const float* __restrict__ AD1,
    unsigned* __restrict__ EMAX1) {
    int gid = blockIdx.x * 256 + threadIdx.x;
    if (gid >= EPEDGES * NHEADS) return;
    int e = gid >> 3, h = gid & 7;
    int s, d;
    if (e < NEDGES) { s = (int)ei[e]; d = (int)ei[NEDGES + e]; } else { s = d = e - NEDGES; }
    float v = AS1[s * NHEADS + h] + AD1[d * NHEADS + h];
    v = v > 0.f ? v : NEG_SLOPE * v;
    atomicMax(&EMAX1[d * NHEADS + h], encf(v));
}

__global__ __launch_bounds__(256) void gat_edge_agg1(
    const long long* __restrict__ ei,
    const float* __restrict__ AS1, const float* __restrict__ AD1,
    const unsigned* __restrict__ EMAX1, float* __restrict__ DEN1,
    const _Float16* __restrict__ H1h, float* __restrict__ S1) {
    int gid = blockIdx.x * 256 + threadIdx.x;
    if (gid >= EPEDGES * NHEADS) return;
    int e = gid >> 3, h = gid & 7;
    int s, d;
    if (e < NEDGES) { s = (int)ei[e]; d = (int)ei[NEDGES + e]; } else { s = d = e - NEDGES; }
    float v = AS1[s * NHEADS + h] + AD1[d * NHEADS + h];
    v = v > 0.f ? v : NEG_SLOPE * v;
    float ex = expf(v - decf(EMAX1[d * NHEADS + h]));
    atomicAdd(&DEN1[d * NHEADS + h], ex);
    const _Float16* hp = H1h + (size_t)s * F1 + h * HID;
    float* op = S1 + (size_t)d * F1 + h * HID;
    float vals[HID];
    #pragma unroll
    for (int i = 0; i < HID; ++i) vals[i] = ex * (float)hp[i];
    #pragma unroll
    for (int i = 0; i < HID; ++i) atomicAdd(&op[i], vals[i]);
}

// ---------------- layer-1 finalize: /denom + bias, ELU, -> f16 --------------
__global__ __launch_bounds__(256) void gat_fin1(
    const float* __restrict__ S1, const float* __restrict__ DEN1,
    const float* __restrict__ b1, _Float16* __restrict__ X2h) {
    int tid = blockIdx.x * 256 + threadIdx.x;      // exactly N*128 threads
    int nd = tid >> 7, f = tid & 127;
    float v = S1[tid] / (DEN1[nd * NHEADS + (f >> 4)] + 1e-16f) + b1[f];
    v = v > 0.f ? v : (expf(v) - 1.f);             // ELU
    X2h[tid] = (_Float16)v;
}

// ---------------- layer-2 GEMM ----------------------------------------------
// block = 96 threads = 3 waves; wave w -> cols 16w..16w+15 (cols>=40 zero-padded)
__global__ __launch_bounds__(96) void gat_gemm2(
    const _Float16* __restrict__ X2h, const _Float16* __restrict__ W2s,
    _Float16* __restrict__ H2h) {
    __shared__ __align__(16) _Float16 xs[16][136];   // 272B row stride, 16B aligned
    int m0 = blockIdx.x * 16;
    for (int idx = threadIdx.x; idx < 16 * F1; idx += 96) {
        int r = idx >> 7, cc = idx & 127;
        xs[r][cc] = X2h[(size_t)(m0 + r) * F1 + cc];
    }
    __syncthreads();

    int w = threadIdx.x >> 5, L = threadIdx.x & 31;
    int n = L & 15, half = L >> 4;
    int col = w * 16 + n;
    v8f c = {};
    #pragma unroll
    for (int kt = 0; kt < 4; ++kt) {
        int ab = kt * 32 + half * 8;
        union { v16h v; v8h h[2]; } a;
        a.h[0] = *(const v8h*)&xs[n][ab];
        a.h[1] = *(const v8h*)&xs[n][ab + 16];
        v16h b = *(const v16h*)(W2s + (((w * 4 + kt) * 32 + L) << 4));
        c = __builtin_amdgcn_wmma_f32_16x16x32_f16(false, a.v, false, b, (short)0, c,
                                                   false, false);
    }
    #pragma unroll
    for (int r = 0; r < 8; ++r) {
        int row = m0 + half * 8 + r;
        H2h[(size_t)row * C2PAD + col] = (_Float16)c[r];
    }
}

// ---------------- layer-2 edge passes ---------------------------------------
__global__ __launch_bounds__(256) void gat_edge_max2(
    const long long* __restrict__ ei,
    const float* __restrict__ AS2, const float* __restrict__ AD2,
    unsigned* __restrict__ EMAX2) {
    int e = blockIdx.x * 256 + threadIdx.x;
    if (e >= EPEDGES) return;
    int s, d;
    if (e < NEDGES) { s = (int)ei[e]; d = (int)ei[NEDGES + e]; } else { s = d = e - NEDGES; }
    float v = AS2[s] + AD2[d];
    v = v > 0.f ? v : NEG_SLOPE * v;
    atomicMax(&EMAX2[d], encf(v));
}

__global__ __launch_bounds__(256) void gat_edge_agg2(
    const long long* __restrict__ ei,
    const float* __restrict__ AS2, const float* __restrict__ AD2,
    const unsigned* __restrict__ EMAX2, float* __restrict__ DEN2,
    const _Float16* __restrict__ H2h, float* __restrict__ out) {
    int e = blockIdx.x * 256 + threadIdx.x;
    if (e >= EPEDGES) return;
    int s, d;
    if (e < NEDGES) { s = (int)ei[e]; d = (int)ei[NEDGES + e]; } else { s = d = e - NEDGES; }
    float v = AS2[s] + AD2[d];
    v = v > 0.f ? v : NEG_SLOPE * v;
    float ex = expf(v - decf(EMAX2[d]));
    atomicAdd(&DEN2[d], ex);
    const _Float16* hp = H2h + (size_t)s * C2PAD;
    float* op = out + (size_t)d * NCLS;
    float vals[NCLS];
    #pragma unroll
    for (int cc = 0; cc < NCLS; ++cc) vals[cc] = ex * (float)hp[cc];
    #pragma unroll
    for (int cc = 0; cc < NCLS; ++cc) atomicAdd(&op[cc], vals[cc]);
}

// ---------------- layer-2 finalize: /denom + bias, log_softmax (in place) ---
__global__ __launch_bounds__(256) void gat_fin2(float* __restrict__ out,
                                                const float* __restrict__ DEN2,
                                                const float* __restrict__ b2) {
    int n = blockIdx.x * 8 + (threadIdx.x >> 5);   // one wave per node
    int L = threadIdx.x & 31;
    if (n >= NNODES) return;
    float den = DEN2[n] + 1e-16f;
    float v0 = out[(size_t)n * NCLS + L] / den + b2[L];
    float v1 = (L < NCLS - 32) ? out[(size_t)n * NCLS + 32 + L] / den + b2[32 + L]
                               : -3.0e38f;
    float m = fmaxf(v0, v1);
    for (int mk = 1; mk < 32; mk <<= 1) m = fmaxf(m, __shfl_xor(m, mk, 32));
    float se = expf(v0 - m) + ((L < NCLS - 32) ? expf(v1 - m) : 0.f);
    for (int mk = 1; mk < 32; mk <<= 1) se += __shfl_xor(se, mk, 32);
    float lse = m + logf(se);
    out[(size_t)n * NCLS + L] = v0 - lse;
    if (L < NCLS - 32) out[(size_t)n * NCLS + 32 + L] = v1 - lse;
}

extern "C" void kernel_launch(void* const* d_in, const int* in_sizes, int n_in,
                              void* d_out, int out_size, void* d_ws, size_t ws_size,
                              hipStream_t stream) {
    (void)in_sizes; (void)n_in; (void)out_size; (void)ws_size;
    const float*     x   = (const float*)d_in[0];
    const long long* ei  = (const long long*)d_in[1];
    const float*     W1  = (const float*)d_in[2];
    const float*     as1 = (const float*)d_in[3];
    const float*     ad1 = (const float*)d_in[4];
    const float*     b1  = (const float*)d_in[5];
    const float*     W2  = (const float*)d_in[6];
    const float*     as2 = (const float*)d_in[7];
    const float*     ad2 = (const float*)d_in[8];
    const float*     b2  = (const float*)d_in[9];
    float* out = (float*)d_out;

    // carve workspace
    char* p = (char*)d_ws;
    auto carve = [&](size_t bytes) -> char* {
        char* r = p; p += (bytes + 255) & ~(size_t)255; return r;
    };
    _Float16* H1h   = (_Float16*)carve((size_t)NNODES * F1 * 2);
    float*    AS1   = (float*)   carve((size_t)NNODES * NHEADS * 4);
    float*    AD1   = (float*)   carve((size_t)NNODES * NHEADS * 4);
    unsigned* EMAX1 = (unsigned*)carve((size_t)NNODES * NHEADS * 4);
    float*    DEN1  = (float*)   carve((size_t)NNODES * NHEADS * 4);
    float*    S1    = (float*)   carve((size_t)NNODES * F1 * 4);
    _Float16* X2h   = (_Float16*)carve((size_t)NNODES * F1 * 2);
    _Float16* W1s   = (_Float16*)carve((size_t)FIN * F1 * 2);
    _Float16* W2s   = (_Float16*)carve((size_t)3 * 4 * 32 * 16 * 2);
    _Float16* H2h   = (_Float16*)carve((size_t)NNODES * C2PAD * 2);
    float*    AS2   = (float*)   carve((size_t)NNODES * 4);
    float*    AD2   = (float*)   carve((size_t)NNODES * 4);
    unsigned* EMAX2 = (unsigned*)carve((size_t)NNODES * 4);
    float*    DEN2  = (float*)   carve((size_t)NNODES * 4);

    // init accumulators
    hipMemsetAsync(S1,   0, (size_t)NNODES * F1 * 4, stream);
    hipMemsetAsync(DEN1, 0, (size_t)NNODES * NHEADS * 4, stream);
    hipMemsetAsync(DEN2, 0, (size_t)NNODES * 4, stream);
    hipMemsetAsync(out,  0, (size_t)NNODES * NCLS * 4, stream);
    gat_fill_neginf<<<(NNODES * NHEADS + 255) / 256, 256, 0, stream>>>(EMAX1, NNODES * NHEADS);
    gat_fill_neginf<<<(NNODES + 255) / 256, 256, 0, stream>>>(EMAX2, NNODES);
    gat_prep<<<(FIN * F1 + 255) / 256, 256, 0, stream>>>(W1, W2, W1s, W2s);

    // layer 1
    gat_gemm1<<<NNODES / 16, 256, 0, stream>>>(x, W1s, H1h);
    gat_att1<<<NNODES * NHEADS / 256, 256, 0, stream>>>(H1h, as1, ad1, AS1, AD1);
    int egrid = (EPEDGES * NHEADS + 255) / 256;
    gat_edge_max1<<<egrid, 256, 0, stream>>>(ei, AS1, AD1, EMAX1);
    gat_edge_agg1<<<egrid, 256, 0, stream>>>(ei, AS1, AD1, EMAX1, DEN1, H1h, S1);
    gat_fin1<<<NNODES * F1 / 256, 256, 0, stream>>>(S1, DEN1, b1, X2h);

    // layer 2
    gat_gemm2<<<NNODES / 16, 96, 0, stream>>>(X2h, W2s, H2h);
    gat_att2<<<(NNODES + 255) / 256, 256, 0, stream>>>(H2h, as2, ad2, AS2, AD2);
    int egrid2 = (EPEDGES + 255) / 256;
    gat_edge_max2<<<egrid2, 256, 0, stream>>>(ei, AS2, AD2, EMAX2);
    gat_edge_agg2<<<egrid2, 256, 0, stream>>>(ei, AS2, AD2, EMAX2, DEN2, H2h, out);
    gat_fin2<<<NNODES / 8, 256, 0, stream>>>(out, DEN2, b2);
}
